// VisionTritonAttention_48069273977254
// MI455X (gfx1250) — matrix operand order repaired and checked by reading, not compiled, not run on hardware
//
#include <hip/hip_runtime.h>
#include <hip/hip_bf16.h>

// CDNA5 / gfx1250 varlen flash attention, f16 WMMA with fp32 accumulation.
typedef __attribute__((ext_vector_type(16))) _Float16 v16h;
typedef __attribute__((ext_vector_type(8)))  _Float16 v8h;
typedef __attribute__((ext_vector_type(4)))  _Float16 v4h;
typedef __attribute__((ext_vector_type(8)))  float    v8f;

#define H_ 16
#define D_ 128

__device__ __forceinline__ v16h ld_b16x16(const _Float16* p) {
  v8h a = *(const v8h*)p;
  v8h b = *(const v8h*)(p + 16);
  v16h r;
#pragma unroll
  for (int j = 0; j < 8; ++j) { r[j] = a[j]; r[j + 8] = b[j]; }
  return r;
}

__global__ __launch_bounds__(128) void fa_fwd_gfx1250(
    const float* __restrict__ Q, const float* __restrict__ K,
    const float* __restrict__ V, const int* __restrict__ cu,
    float* __restrict__ O) {
  constexpr int KR = D_ + 8;  // Kld row stride (halves): 136 -> 272B, conflict-free
  constexpr int VR = 40;      // Vt row stride  (halves): 80B, 16B aligned
  constexpr int PR = 40;      // P  row stride  (halves)
  __shared__ _Float16 Kld[32 * KR];     // K tile: [key][dim], f16
  __shared__ _Float16 Vt[D_ * VR];      // V tile transposed: [dim][key], f16
  __shared__ _Float16 Pld[4 * 16 * PR]; // per-wave P bounce: [row][key]

  const int tid  = threadIdx.x;
  const int wave = tid >> 5, lane = tid & 31;
  const int l16  = lane & 15;
  const bool hi  = lane >= 16;

  const int b  = blockIdx.y, h = blockIdx.z;
  const int s0 = cu[b];
  const int L  = cu[b + 1] - s0;
  const int q0 = blockIdx.x * 64;
  if (q0 >= L) return;  // whole block exits together

  const float scale = 0.08838834764831845f;  // 1/sqrt(128)

  // ---- Load Q fragments (A-layout, 16x32 f16 per K-step; 4 steps for D=128)
  // A 16x32 layout: lanes 0-15 hold M=l16, K={0..7,16..23}; lanes 16-31 K={8..15,24..31}
  const int qrow   = q0 + wave * 16 + l16;
  const int qclamp = qrow < L ? qrow : (L - 1);
  const float* qp  = Q + ((size_t)(s0 + qclamp) * H_ + h) * D_;
  v16h aq[4];
#pragma unroll
  for (int f = 0; f < 4; ++f) {
    int base = f * 32 + (hi ? 8 : 0);
    float4 x0 = *(const float4*)(qp + base);
    float4 x1 = *(const float4*)(qp + base + 4);
    float4 y0 = *(const float4*)(qp + base + 16);
    float4 y1 = *(const float4*)(qp + base + 20);
    float t[16] = {x0.x, x0.y, x0.z, x0.w, x1.x, x1.y, x1.z, x1.w,
                   y0.x, y0.y, y0.z, y0.w, y1.x, y1.y, y1.z, y1.w};
#pragma unroll
    for (int j = 0; j < 16; ++j) aq[f][j] = (_Float16)(t[j] * scale);
  }

  // ---- Online-softmax state. C-layout: VGPR r -> row r (lanes 0-15) / row 8+r (16-31)
  float m[8], lsum[8], alpha[8];
  v8f o[8];
#pragma unroll
  for (int r = 0; r < 8; ++r) { m[r] = -1e30f; lsum[r] = 0.f; }
#pragma unroll
  for (int dt = 0; dt < 8; ++dt)
#pragma unroll
    for (int j = 0; j < 8; ++j) o[dt][j] = 0.f;

  _Float16* myP = Pld + wave * 16 * PR;

  for (int kb = 0; kb < L; kb += 32) {
    __syncthreads();  // protect LDS tiles from previous iteration's readers
    // ---- Stage K [32][128] and V^T [128][32] (fp32 global -> f16 LDS)
#pragma unroll
    for (int it = 0; it < 8; ++it) {
      int i4  = tid + it * 128;      // 1024 float4 tiles total
      int row = i4 >> 5;             // key row 0..31
      int d4  = (i4 & 31) << 2;      // dim 0..124 step 4
      int tok = kb + row;
      float4 kx, vx;
      if (tok < L) {
        size_t base = ((size_t)(s0 + tok) * H_ + h) * D_ + d4;
        kx = *(const float4*)(K + base);
        vx = *(const float4*)(V + base);
        if (tok + 32 < L) {  // gfx1250 global_prefetch_b8 of next key block
          __builtin_prefetch(K + base + (size_t)32 * H_ * D_, 0, 0);
          __builtin_prefetch(V + base + (size_t)32 * H_ * D_, 0, 0);
        }
      } else {
        kx = make_float4(0.f, 0.f, 0.f, 0.f);
        vx = make_float4(0.f, 0.f, 0.f, 0.f);
      }
      v4h kh;
      kh[0] = (_Float16)kx.x; kh[1] = (_Float16)kx.y;
      kh[2] = (_Float16)kx.z; kh[3] = (_Float16)kx.w;
      *(v4h*)(Kld + row * KR + d4) = kh;
      Vt[(d4 + 0) * VR + row] = (_Float16)vx.x;
      Vt[(d4 + 1) * VR + row] = (_Float16)vx.y;
      Vt[(d4 + 2) * VR + row] = (_Float16)vx.z;
      Vt[(d4 + 3) * VR + row] = (_Float16)vx.w;
    }
    __syncthreads();

    // ---- S = Q K^T : two 16x16 tiles (keys kb..kb+15, kb+16..kb+31)
    // B 32x16 layout: lane = column (key), K(dim) halves {0..7,16..23} per lane,
    // lanes 16-31 hold dims +8.
    v8f c0, c1;
#pragma unroll
    for (int j = 0; j < 8; ++j) { c0[j] = 0.f; c1[j] = 0.f; }
#pragma unroll
    for (int f = 0; f < 4; ++f) {
      int dbase = f * 32 + (hi ? 8 : 0);
      v16h bk0 = ld_b16x16(Kld + l16 * KR + dbase);
      v16h bk1 = ld_b16x16(Kld + (16 + l16) * KR + dbase);
      c0 = __builtin_amdgcn_wmma_f32_16x16x32_f16(false, aq[f], false, bk0,
                                                  (short)0, c0, false, false);
      c1 = __builtin_amdgcn_wmma_f32_16x16x32_f16(false, aq[f], false, bk1,
                                                  (short)0, c1, false, false);
    }

    // ---- Mask out-of-range keys
    if (kb + l16 >= L) {
#pragma unroll
      for (int r = 0; r < 8; ++r) c0[r] = -1e30f;
    }
    if (kb + 16 + l16 >= L) {
#pragma unroll
      for (int r = 0; r < 8; ++r) c1[r] = -1e30f;
    }

    // ---- Online softmax: row reductions within each 16-lane half
#pragma unroll
    for (int r = 0; r < 8; ++r) {
      float mx = fmaxf(c0[r], c1[r]);
      mx = fmaxf(mx, __shfl_xor(mx, 1));
      mx = fmaxf(mx, __shfl_xor(mx, 2));
      mx = fmaxf(mx, __shfl_xor(mx, 4));
      mx = fmaxf(mx, __shfl_xor(mx, 8));
      float mn = fmaxf(m[r], mx);
      alpha[r] = __expf(m[r] - mn);
      m[r] = mn;
      float p0 = __expf(c0[r] - mn);
      float p1 = __expf(c1[r] - mn);
      int prow = r + (hi ? 8 : 0);
      myP[prow * PR + l16]      = (_Float16)p0;  // P bounce: C-layout -> LDS
      myP[prow * PR + 16 + l16] = (_Float16)p1;
      float s = p0 + p1;
      s += __shfl_xor(s, 1);
      s += __shfl_xor(s, 2);
      s += __shfl_xor(s, 4);
      s += __shfl_xor(s, 8);
      lsum[r] = lsum[r] * alpha[r] + s;
    }

    // ---- Rescale accumulator
#pragma unroll
    for (int dt = 0; dt < 8; ++dt)
#pragma unroll
      for (int r = 0; r < 8; ++r) o[dt][r] *= alpha[r];

    // ---- P as A-fragment (16 rows x 32 keys); same-wave LDS is in-order
    v16h ap;
    {
      int kbase = hi ? 8 : 0;
      v8h x0 = *(const v8h*)(myP + l16 * PR + kbase);
      v8h x1 = *(const v8h*)(myP + l16 * PR + kbase + 16);
#pragma unroll
      for (int j = 0; j < 8; ++j) { ap[j] = x0[j]; ap[j + 8] = x1[j]; }
    }

    // ---- O += P V : 8 dim-tiles, V^T rows give B fragments of two 8-key runs
#pragma unroll
    for (int dt = 0; dt < 8; ++dt) {
      int dim   = dt * 16 + l16;
      int kbase = hi ? 8 : 0;
      v16h bv = ld_b16x16(Vt + dim * VR + kbase);
      o[dt] = __builtin_amdgcn_wmma_f32_16x16x32_f16(false, ap, false, bv,
                                                     (short)0, o[dt], false, false);
    }
  }

  // ---- Epilogue: out[row][h][dim] = o / l  (half-wave writes 64B runs)
#pragma unroll
  for (int r = 0; r < 8; ++r) {
    int row = q0 + wave * 16 + r + (hi ? 8 : 0);
    if (row < L) {
      float inv = 1.0f / lsum[r];
      float* op = O + ((size_t)(s0 + row) * H_ + h) * D_ + l16;
#pragma unroll
      for (int dt = 0; dt < 8; ++dt) op[dt * 16] = o[dt][r] * inv;
    }
  }
}

extern "C" void kernel_launch(void* const* d_in, const int* in_sizes, int n_in,
                              void* d_out, int out_size, void* d_ws, size_t ws_size,
                              hipStream_t stream) {
  const float* q  = (const float*)d_in[0];
  const float* k  = (const float*)d_in[1];
  const float* v  = (const float*)d_in[2];
  const int*   cu = (const int*)d_in[3];
  float*       out = (float*)d_out;

  int nseq = in_sizes[3] - 1;          // 8 sequences
  // MAX_S = 1024 -> at most 16 q-tiles of 64 rows; extras early-exit on device.
  dim3 grid(16, nseq, H_);
  fa_fwd_gfx1250<<<grid, 128, 0, stream>>>(q, k, v, cu, out);
}
